// CustomFullyConnectedLayer_90288802497264
// MI455X (gfx1250) — matrix operand order repaired and checked by reading, not compiled, not run on hardware
//
#include <hip/hip_runtime.h>

typedef __attribute__((ext_vector_type(16))) __bf16 v16bf;
typedef __attribute__((ext_vector_type(8)))  __bf16 bf16x8;
typedef __attribute__((ext_vector_type(8)))  float  v8f;
typedef __attribute__((ext_vector_type(4)))  float  f32x4;

#define IN_F   768
#define OUT_F  2304
#define PERM   2304
#define ROWS   16384      // 16 * 1024
#define KSEL   231
#define NITER  50
#define EPW    9          // 2304 / 256 elements per thread for Dykstra
#define NT     8          // N-tiles of 16 per wave (wave tile = 16M x 128N)

// ---------------------------------------------------------------------------
// Kernel 1: Dykstra projection of z = alpha/0.01 onto {0<=m<=1, sum(m)=K}.
// One workgroup; y,q live in registers; p is a uniform scalar (the hyperplane
// correction is constant across elements); block-wide sum via LDS tree.
// ---------------------------------------------------------------------------
__global__ __launch_bounds__(256)
void dykstra_mask(const float* __restrict__ alpha, float* __restrict__ m)
{
    __shared__ float red[256];
    const int tid = threadIdx.x;

    float y[EPW], q[EPW];
    float p = 0.0f;
    #pragma unroll
    for (int e = 0; e < EPW; ++e) {
        y[e] = alpha[tid * EPW + e] * 100.0f;   // alpha / ALPHA_LR
        q[e] = 0.0f;
    }

    for (int it = 0; it < NITER; ++it) {
        float v[EPW];
        float local = 0.0f;
        #pragma unroll
        for (int e = 0; e < EPW; ++e) { v[e] = y[e] + p; local += v[e]; }

        red[tid] = local;
        __syncthreads();
        #pragma unroll
        for (int s = 128; s > 0; s >>= 1) {
            if (tid < s) red[tid] += red[tid + s];
            __syncthreads();
        }
        const float c = ((float)KSEL - red[0]) * (1.0f / (float)PERM);
        __syncthreads();   // protect red[] before next iteration's write

        p = -c;            // p_new = v - v_h = -c (uniform)
        #pragma unroll
        for (int e = 0; e < EPW; ++e) {
            float w  = v[e] + c + q[e];
            float yn = fminf(fmaxf(w, 0.0f), 1.0f);
            q[e] = w - yn;
            y[e] = yn;
        }
    }

    #pragma unroll
    for (int e = 0; e < EPW; ++e) m[tid * EPW + e] = y[e];
}

// ---------------------------------------------------------------------------
// Kernel 2: W[o,j] = m[(o-j) mod 2304] * V[(o-j) mod 2304, j], split into
// bf16 hi + bf16 lo so that hi + lo ~= fp32 value.
// ---------------------------------------------------------------------------
__global__ __launch_bounds__(256)
void build_w(const float* __restrict__ V, const float* __restrict__ m,
             __bf16* __restrict__ Whi, __bf16* __restrict__ Wlo)
{
    const int idx = blockIdx.x * 256 + threadIdx.x;
    if (idx >= OUT_F * IN_F) return;
    const int o = idx / IN_F;
    const int j = idx - o * IN_F;
    int i = o - j;
    if (i < 0) i += PERM;

    const float w = m[i] * V[(size_t)i * IN_F + j];
    const __bf16 h = (__bf16)w;
    Whi[idx] = h;
    Wlo[idx] = (__bf16)(w - (float)h);
}

// ---------------------------------------------------------------------------
// Kernel 3: out[t,o] = sum_j x[t,j] * W[o,j] via bf16 WMMA with hi/lo split
// (x_hi*W_hi + x_hi*W_lo + x_lo*W_hi), fp32 accumulation.
// Block = 256 threads = 8 wave32. Wave computes a 16(M) x 128(N) tile:
// the A-tile load + fp32->bf16 hi/lo split is amortized over 8 B-tiles
// (24 WMMAs per K-step vs ~32 convert-VALU).
//
// WMMA VGPR layouts (ISA 7.12.2, wave32):
//  A (16x32 bf16): lane L -> row M = L%16; halves 0..7 = K kb+(L<16?0:8)+0..7,
//                  halves 8..15 = K kb+16+(L<16?0:8)+0..7.
//  B (32x16 bf16): lane L -> col N = L%16; halves 0..7 = K kb+(L<16?0:8)+0..7, 8..15 = K kb+16+(L<16?0:8)+0..7.
//  C/D (16x16 f32): lane L -> col N = L%16; VGPR r -> row M = r + (L<16?0:8).
// ---------------------------------------------------------------------------
__global__ __launch_bounds__(256)
void gemm_bf16x3(const float* __restrict__ x,
                 const __bf16* __restrict__ Whi,
                 const __bf16* __restrict__ Wlo,
                 float* __restrict__ out)
{
    const int lane = threadIdx.x & 31;
    const int wave = threadIdx.x >> 5;
    const int half = lane >> 4;     // 0: lanes 0-15, 1: lanes 16-31
    const int l16  = lane & 15;

    const int tM = (blockIdx.y * 8 + wave) * 16;   // M-tile base (rows of x)
    const int nN = blockIdx.x * (NT * 16);         // N base (cols of out)

    v8f acc[NT] = {};

    const float* xrow = x + (size_t)(tM + l16) * IN_F + half * 8;

    for (int kb = 0; kb < IN_F; kb += 32) {
        // ---- A tile: load fp32, split into bf16 hi/lo in registers ----
        f32x4 a0 = *(const f32x4*)(xrow + kb + 0);
        f32x4 a1 = *(const f32x4*)(xrow + kb + 4);
        f32x4 a2 = *(const f32x4*)(xrow + kb + 16);
        f32x4 a3 = *(const f32x4*)(xrow + kb + 20);

        float av[16];
        #pragma unroll
        for (int e = 0; e < 4; ++e) {
            av[e]      = a0[e];
            av[4 + e]  = a1[e];
            av[8 + e]  = a2[e];
            av[12 + e] = a3[e];
        }

        v16bf a_hi, a_lo;
        #pragma unroll
        for (int e = 0; e < 16; ++e) {
            __bf16 h = (__bf16)av[e];
            a_hi[e] = h;
            a_lo[e] = (__bf16)(av[e] - (float)h);
        }

        // ---- 8 B tiles; 3 WMMAs each (hi*hi, hi*lo, lo*hi) ----
        #pragma unroll
        for (int nt = 0; nt < NT; ++nt) {
            const size_t wofs = (size_t)(nN + nt * 16 + l16) * IN_F + kb + half * 8;
            bf16x8 bh0 = *(const bf16x8*)(Whi + wofs);
            bf16x8 bh1 = *(const bf16x8*)(Whi + wofs + 16);
            bf16x8 bl0 = *(const bf16x8*)(Wlo + wofs);
            bf16x8 bl1 = *(const bf16x8*)(Wlo + wofs + 16);

            v16bf b_hi, b_lo;
            #pragma unroll
            for (int e = 0; e < 8; ++e) {
                b_hi[e]     = bh0[e];
                b_hi[e + 8] = bh1[e];
                b_lo[e]     = bl0[e];
                b_lo[e + 8] = bl1[e];
            }

            acc[nt] = __builtin_amdgcn_wmma_f32_16x16x32_bf16(
                false, a_hi, false, b_hi, (short)0, acc[nt], false, false);
            acc[nt] = __builtin_amdgcn_wmma_f32_16x16x32_bf16(
                false, a_hi, false, b_lo, (short)0, acc[nt], false, false);
            acc[nt] = __builtin_amdgcn_wmma_f32_16x16x32_bf16(
                false, a_lo, false, b_hi, (short)0, acc[nt], false, false);
        }
    }

    // ---- store D (fp32): lane col = l16, VGPR r -> row half*8 + r ----
    #pragma unroll
    for (int nt = 0; nt < NT; ++nt) {
        const int ocol = nN + nt * 16 + l16;
        #pragma unroll
        for (int r = 0; r < 8; ++r) {
            out[(size_t)(tM + half * 8 + r) * OUT_F + ocol] = acc[nt][r];
        }
    }
}

// ---------------------------------------------------------------------------
extern "C" void kernel_launch(void* const* d_in, const int* in_sizes, int n_in,
                              void* d_out, int out_size, void* d_ws, size_t ws_size,
                              hipStream_t stream)
{
    (void)in_sizes; (void)n_in; (void)out_size; (void)ws_size;

    const float* x     = (const float*)d_in[0];   // [16,1024,768]
    const float* V     = (const float*)d_in[1];   // [2304,768]
    const float* alpha = (const float*)d_in[2];   // [2304]
    float* out = (float*)d_out;                   // [16,1024,2304]

    char* ws = (char*)d_ws;
    float*  m   = (float*)ws;                                       // 9216 B
    __bf16* Whi = (__bf16*)(ws + 16384);                            // 3.54 MB
    __bf16* Wlo = (__bf16*)(ws + 16384 + (size_t)OUT_F * IN_F * 2); // 3.54 MB

    dykstra_mask<<<1, 256, 0, stream>>>(alpha, m);
    build_w<<<(OUT_F * IN_F + 255) / 256, 256, 0, stream>>>(V, m, Whi, Wlo);

    dim3 grid(OUT_F / (NT * 16), ROWS / 128);   // (18, 128)
    gemm_bf16x3<<<grid, 256, 0, stream>>>(x, Whi, Wlo, out);
}
